// SGN_74904229642657
// MI455X (gfx1250) — compile-verified, matrix operand order checked by reading, not run on hardware
//
#include <hip/hip_runtime.h>

typedef unsigned short u16;
typedef __attribute__((ext_vector_type(4)))  unsigned short u16x4;
typedef __attribute__((ext_vector_type(8)))  float  v8f;
typedef __attribute__((ext_vector_type(8)))  __bf16 v8bf;
typedef __attribute__((ext_vector_type(16))) __bf16 v16bf;

#define KBS 256
#define KTJ 500
#define EM_RELU 1
#define EM_BIAS 2
#define EM_BN   4
#define EM_ACC  8

__device__ __forceinline__ u16 f2bf(float f) {
  union { float f; unsigned u; } v; v.f = f;
  unsigned r = v.u + 0x7FFFu + ((v.u >> 16) & 1u);
  return (u16)(r >> 16);
}
__device__ __forceinline__ float bf2f(u16 h) {
  union { unsigned u; float f; } v; v.u = ((unsigned)h) << 16;
  return v.f;
}

// ---------------------------------------------------------------------------
// Generic batched bf16 WMMA GEMM:  C[b](M,N) = A[b](M,K) * B[b](K,N)
// A element (m,k)  at A + b*a_bs + m*sam + k*sak   (fast path assumes sak==1)
// B element (k,n)  from B0 if k<K1 else B1 (k-K1); strides sbk,sbn shared.
//   sbk==1 : K-contiguous  -> LDS B stored [n][k], frags = 2x ds_load_b128
//   sbn==1 : N-contiguous  -> LDS B stored [k][n], frags = 16x ds_load_u16
// Tile: 64(M) x 128(N) x 32(K); 8 waves, each wave 2x2 16x16 subtiles ->
// 4 v_wmma_f32_16x16x32_bf16 per wave per K step.
// Epilogue (per m): +bias, *bnsc+bnof, relu; bf16 store or f32 accumulate
// (alpha * chsc[m] * val).
// ---------------------------------------------------------------------------
__global__ __launch_bounds__(256) void sgn_gemm(
    const u16* __restrict__ A, long a_bs, int sam, int sak,
    const u16* __restrict__ B0, long b0_bs,
    const u16* __restrict__ B1, long b1_bs,
    int sbk, int sbn, int K1, int K, int M, int N,
    const float* __restrict__ bias, const float* __restrict__ bnsc,
    const float* __restrict__ bnof, const float* __restrict__ chsc,
    float alpha, int emode,
    u16* __restrict__ outb, float* __restrict__ outf,
    long o_bs, int som, int son)
{
  __shared__ __align__(16) u16 As[64 * 40];    // [m][k], row stride 40
  __shared__ __align__(16) u16 Bsh[128 * 40];  // sbk==1: [n][k] stride 40
                                               // else  : [k][n] stride 136
  const int b = blockIdx.z;
  const int mBase = blockIdx.y << 6;
  const int nBase = blockIdx.x << 7;
  const int t = threadIdx.x;
  const u16* Ab  = A  + (long)b * a_bs;
  const u16* B0b = B0 + (long)b * b0_bs;
  const u16* B1b = B1 + (long)b * b1_bs;
  const bool bk1 = (sbk == 1);

  const int w = t >> 5, lane = t & 31;
  const int wm = w >> 2, wn = w & 3;
  const int lo = lane & 15, hi = lane >> 4;

  v8f acc[2][2] = {};

  const int ksteps = (K + 31) >> 5;
  for (int kb = 0; kb < ksteps; ++kb) {
    const int k0 = kb << 5;
    if (kb + 1 < ksteps) {   // speculative next-tile prefetch (global_prefetch_b8)
      const int kn = k0 + 32;
      __builtin_prefetch(Ab + (long)(mBase + (t & 63)) * sam + (long)kn * sak, 0, 1);
      __builtin_prefetch(B0b + (long)kn * sbk + (long)(nBase + (t & 127)) * sbn, 0, 1);
    }

    // ---- A tile: 64x32, 8 elements/thread, b64 fast path (sak==1) ----
    {
      const int i = t >> 2, kc = (t & 3) << 3;
      const int m = mBase + i;
      u16* dst = &As[i * 40 + kc];
#pragma unroll
      for (int h = 0; h < 2; ++h) {
        const int k = k0 + kc + (h << 2);
        if (sak == 1 && m < M && k + 3 < K) {
          *(u16x4*)(dst + (h << 2)) = *(const u16x4*)(Ab + (long)m * sam + k);
        } else {
#pragma unroll
          for (int q = 0; q < 4; ++q) {
            const int kq = k + q;
            const bool p = (m < M) && (kq < K);
            const long ix = p ? ((long)m * sam + (long)kq * sak) : 0;
            u16 v = Ab[ix];
            dst[(h << 2) + q] = p ? v : (u16)0;
          }
        }
      }
    }

    // ---- B tile: 32x128, 16 elements/thread ----
    if (bk1) {
      // [n][k] layout, K-contiguous global (call sites here have K1==K,
      // K1 multiple of 32 -> no intra-chunk source switch)
      const int j = t >> 1, kc = (t & 1) << 4;
      const int n = nBase + j;
      u16* dst = &Bsh[j * 40 + kc];
#pragma unroll
      for (int h = 0; h < 4; ++h) {
        const int k = k0 + kc + (h << 2);
        const bool s0 = k < K1;
        if (n < N && k + 3 < K && (s0 ? (k + 3 < K1) : true)) {
          const u16* s = s0 ? (B0b + (long)k + (long)n * sbn)
                            : (B1b + (long)(k - K1) + (long)n * sbn);
          *(u16x4*)(dst + (h << 2)) = *(const u16x4*)s;
        } else {
#pragma unroll
          for (int q = 0; q < 4; ++q) {
            const int kq = k + q;
            const bool p = (n < N) && (kq < K);
            const long ix = p ? ((kq < K1) ? ((long)kq * sbk + (long)n * sbn)
                                           : ((long)(kq - K1) * sbk + (long)n * sbn))
                              : 0;
            u16 v = (p && kq >= K1) ? B1b[ix] : B0b[ix];
            dst[(h << 2) + q] = p ? v : (u16)0;
          }
        }
      }
    } else {
      // [k][n] layout (stride 136), N-contiguous global; k uniform per row
      // and K1 multiple of 32 -> whole row from one source
      const int kk = t >> 3, jc = (t & 7) << 4;
      const int k = k0 + kk;
      const bool s0 = k < K1;
      const u16* Bb = s0 ? B0b : B1b;
      const long krow = s0 ? (long)k * sbk : (long)(k - K1) * sbk;
      u16* dst = &Bsh[kk * 136 + jc];
#pragma unroll
      for (int h = 0; h < 4; ++h) {
        const int n = nBase + jc + (h << 2);
        if (k < K && sbn == 1 && n + 3 < N) {
          *(u16x4*)(dst + (h << 2)) = *(const u16x4*)(Bb + krow + n);
        } else {
#pragma unroll
          for (int q = 0; q < 4; ++q) {
            const int nq = n + q;
            const bool p = (k < K) && (nq < N);
            const long ix = p ? (krow + (long)nq * sbn) : 0;
            u16 v = Bb[ix];
            dst[(h << 2) + q] = p ? v : (u16)0;
          }
        }
      }
    }
    __syncthreads();

    // ---- fragments (ISA 16-bit layout: lanes 0-15 K0-7/16-23,
    //      lanes 16-31 K8-15/24-31 -> two contiguous 16B reads) ----
    const int r0 = (wm << 5) + lo;
    v8bf a0l = *(const v8bf*)&As[r0 * 40 + (hi << 3)];
    v8bf a0h = *(const v8bf*)&As[r0 * 40 + 16 + (hi << 3)];
    v8bf a1l = *(const v8bf*)&As[(r0 + 16) * 40 + (hi << 3)];
    v8bf a1h = *(const v8bf*)&As[(r0 + 16) * 40 + 16 + (hi << 3)];
    v16bf af0 = __builtin_shufflevector(a0l, a0h, 0,1,2,3,4,5,6,7,8,9,10,11,12,13,14,15);
    v16bf af1 = __builtin_shufflevector(a1l, a1h, 0,1,2,3,4,5,6,7,8,9,10,11,12,13,14,15);

    v16bf bf01[2];
#pragma unroll
    for (int sn = 0; sn < 2; ++sn) {
      const int cB = (wn << 5) + (sn << 4) + lo;
      if (bk1) {
        v8bf bl = *(const v8bf*)&Bsh[cB * 40 + (hi << 3)];
        v8bf bh = *(const v8bf*)&Bsh[cB * 40 + 16 + (hi << 3)];
        bf01[sn] = __builtin_shufflevector(bl, bh, 0,1,2,3,4,5,6,7,8,9,10,11,12,13,14,15);
      } else {
        union { u16 s16[16]; v16bf v; } u;
#pragma unroll
        for (int e = 0; e < 16; ++e) {
          const int kkE = (e < 8) ? ((hi << 3) + e) : (16 + (hi << 3) + (e - 8));
          u.s16[e] = Bsh[kkE * 136 + cB];
        }
        bf01[sn] = u.v;
      }
    }

    acc[0][0] = __builtin_amdgcn_wmma_f32_16x16x32_bf16(false, af0, false, bf01[0], (short)0, acc[0][0], false, false);
    acc[0][1] = __builtin_amdgcn_wmma_f32_16x16x32_bf16(false, af0, false, bf01[1], (short)0, acc[0][1], false, false);
    acc[1][0] = __builtin_amdgcn_wmma_f32_16x16x32_bf16(false, af1, false, bf01[0], (short)0, acc[1][0], false, false);
    acc[1][1] = __builtin_amdgcn_wmma_f32_16x16x32_bf16(false, af1, false, bf01[1], (short)0, acc[1][1], false, false);
    __syncthreads();
  }

#pragma unroll
  for (int sm = 0; sm < 2; ++sm) {
#pragma unroll
    for (int sn = 0; sn < 2; ++sn) {
      v8f a = acc[sm][sn];
#pragma unroll
      for (int v = 0; v < 8; ++v) {
        const int m = mBase + (wm << 5) + (sm << 4) + (hi << 3) + v;
        const int n = nBase + (wn << 5) + (sn << 4) + lo;
        if (m < M && n < N) {
          float val = a[v];
          if (emode & EM_BIAS) val += bias[m];
          if (emode & EM_BN)   val = val * bnsc[m] + bnof[m];
          if (emode & EM_RELU) val = fmaxf(val, 0.f);
          const long oi = (long)b * o_bs + (long)m * som + (long)n * son;
          if (emode & EM_ACC) outf[oi] += alpha * (chsc ? chsc[m] : 1.f) * val;
          else                outb[oi] = f2bf(val);
        }
      }
    }
  }
}

// ---------------- softmax over last dim (500) of LG, in place, bf16 ---------
__global__ __launch_bounds__(256) void sgn_softmax(u16* __restrict__ LG) {
  long base = ((long)blockIdx.y * KTJ + blockIdx.x) * KTJ;
  u16* row = LG + base;
  __shared__ float red[8];
  const int t = threadIdx.x;
  float m = -3.4e38f;
  for (int i = t; i < KTJ; i += 256) m = fmaxf(m, bf2f(row[i]));
#pragma unroll
  for (int o = 16; o; o >>= 1) m = fmaxf(m, __shfl_xor(m, o, 32));
  if ((t & 31) == 0) red[t >> 5] = m;
  __syncthreads();
  if (t == 0) { float v = red[0]; for (int i = 1; i < 8; ++i) v = fmaxf(v, red[i]); red[0] = v; }
  __syncthreads();
  const float rmax = red[0];
  __syncthreads();
  float s = 0.f;
  for (int i = t; i < KTJ; i += 256) s += __expf(bf2f(row[i]) - rmax);
#pragma unroll
  for (int o = 16; o; o >>= 1) s += __shfl_xor(s, o, 32);
  if ((t & 31) == 0) red[t >> 5] = s;
  __syncthreads();
  if (t == 0) { float v = 0.f; for (int i = 0; i < 8; ++i) v += red[i]; red[0] = v; }
  __syncthreads();
  const float inv = 1.f / red[0];
  for (int i = t; i < KTJ; i += 256)
    row[i] = f2bf(__expf(bf2f(row[i]) - rmax) * inv);
}

// ---------------- small / elementwise kernels ------------------------------
__global__ void sgn_f2bf(const float* __restrict__ s, u16* __restrict__ d, int n) {
  int i = blockIdx.x * 256 + threadIdx.x;
  if (i < n) d[i] = f2bf(s[i]);
}
__global__ void sgn_zero(float* __restrict__ p, int n) {
  int i = blockIdx.x * 256 + threadIdx.x;
  if (i < n) p[i] = 0.f;
}
// dst[o][0..Ci-1]=w, dst[o][Ci..2Ci-1]=w1   (fused GCN weight)
__global__ void sgn_pack(const float* __restrict__ w, const float* __restrict__ w1,
                         u16* __restrict__ dst, int O, int Ci) {
  int i = blockIdx.x * 256 + threadIdx.x;
  if (i >= O * Ci) return;
  int o = i / Ci, c = i % Ci;
  dst[(long)o * 2 * Ci + c]      = f2bf(w[i]);
  dst[(long)o * 2 * Ci + Ci + c] = f2bf(w1[i]);
}
__global__ void sgn_bnfold(const float* __restrict__ g, const float* __restrict__ b,
                           const float* __restrict__ m, const float* __restrict__ v,
                           int n, float* __restrict__ sc, float* __restrict__ of) {
  int i = blockIdx.x * 64 + threadIdx.x;
  if (i >= n) return;
  float s = g[i] * rsqrtf(v[i] + 1e-5f);
  sc[i] = s;
  of[i] = b[i] - m[i] * s;
}
// data_bn + reorganize + temporal diff.  Lin/dif: [b][6][500], node=j*20+t
__global__ void sgn_prep(const float* __restrict__ x,
                         const float* __restrict__ gb, const float* __restrict__ gg,
                         const float* __restrict__ gm, const float* __restrict__ gv,
                         float* __restrict__ Lin, float* __restrict__ dif) {
  int i = blockIdx.x * 256 + threadIdx.x;
  if (i >= KBS * KTJ * 6) return;
  int c = i % 6, node = (i / 6) % KTJ, b = i / (6 * KTJ);
  int j = node / 20, tt = node % 20;
  int ci = j * 6 + c;
  float s = gg[ci] * rsqrtf(gv[ci] + 1e-5f);
  float cur = (x[((long)b * 20 + tt) * 150 + ci] - gm[ci]) * s + gb[ci];
  float d = 0.f;
  if (tt > 0) {
    float prev = (x[((long)b * 20 + tt - 1) * 150 + ci] - gm[ci]) * s + gb[ci];
    d = cur - prev;
  }
  Lin[((long)b * 6 + c) * KTJ + node] = cur;
  dif[((long)b * 6 + c) * KTJ + node] = d;
}
// first embed layer: Y[b,64,n] = relu(W(64x6) X[b,6,n] + b)
__global__ void sgn_embed1(const float* __restrict__ X, const float* __restrict__ W1,
                           const float* __restrict__ B1, float* __restrict__ Y) {
  int i = blockIdx.x * 256 + threadIdx.x;
  if (i >= KBS * 64 * KTJ) return;
  int n = i % KTJ, o = (i / KTJ) % 64, b = i / (64 * KTJ);
  float a = B1[o];
#pragma unroll
  for (int c = 0; c < 6; ++c) a += W1[o * 6 + c] * X[((long)b * 6 + c) * KTJ + n];
  Y[((long)b * 64 + o) * KTJ + n] = fmaxf(a, 0.f);
}
// Linput[:, 0..63] = relu(dj2(Epos)) + relu(dd2(Edif))
__global__ void sgn_linput(const float* __restrict__ Ep, const float* __restrict__ Ed,
                           const float* __restrict__ wj, const float* __restrict__ bj,
                           const float* __restrict__ wd, const float* __restrict__ bd,
                           u16* __restrict__ Lout) {
  int i = blockIdx.x * 256 + threadIdx.x;
  if (i >= KBS * 64 * KTJ) return;
  int n = i % KTJ, o = (i / KTJ) % 64, b = i / (64 * KTJ);
  float p = bj[o], d = bd[o];
  for (int c = 0; c < 64; ++c) {
    float ev = Ep[((long)b * 64 + c) * KTJ + n];
    float dv = Ed[((long)b * 64 + c) * KTJ + n];
    p += wj[o * 64 + c] * ev;
    d += wd[o * 64 + c] * dv;
  }
  Lout[((long)b * 128 + o) * KTJ + n] = f2bf(fmaxf(p, 0.f) + fmaxf(d, 0.f));
}
// Linput[:, 64..127] = spa1 (batch-invariant broadcast)
__global__ void sgn_bcast(const float* __restrict__ spa1, u16* __restrict__ Lout) {
  int i = blockIdx.x * 256 + threadIdx.x;
  if (i >= KBS * 64 * KTJ) return;
  int n = i % KTJ, o = (i / KTJ) % 64, b = i / (64 * KTJ);
  Lout[((long)b * 128 + 64 + o) * KTJ + n] = f2bf(spa1[o * KTJ + n]);
}
// one-hot first layer collapses to a lookup: out[o,n]=relu(W1[o,sel(n)]+b1[o])
__global__ void sgn_onehot(const float* __restrict__ W1, const float* __restrict__ B1,
                           float* __restrict__ out, int S, int usemod) {
  int i = blockIdx.x * 256 + threadIdx.x;
  if (i >= 64 * KTJ) return;
  int n = i % KTJ, o = i / KTJ;
  int sel = usemod ? (n % 20) : (n / 20);
  out[o * KTJ + n] = fmaxf(W1[o * S + sel] + B1[o], 0.f);
}
// dense 64 -> O over 500 nodes (batch-invariant)
__global__ void sgn_dense2(const float* __restrict__ X, const float* __restrict__ W2,
                           const float* __restrict__ B2, float* __restrict__ out, int O) {
  int i = blockIdx.x * 256 + threadIdx.x;
  if (i >= O * KTJ) return;
  int n = i % KTJ, o = i / KTJ;
  float a = B2[o];
  for (int c = 0; c < 64; ++c) a += W2[o * 64 + c] * X[c * KTJ + n];
  out[o * KTJ + n] = fmaxf(a, 0.f);
}
// Lf = (accum + off34)/3 + tem1; y0[b,c,t] = max_j Lf
__global__ void sgn_combmax(const float* __restrict__ acc, const float* __restrict__ o34,
                            const float* __restrict__ tem1, float* __restrict__ y0) {
  int i = blockIdx.x * 256 + threadIdx.x;
  if (i >= KBS * 256 * 20) return;
  int t = i % 20, c = (i / 20) % 256, b = i / (256 * 20);
  float best = -3.4e38f;
  const float* ap = acc + ((long)b * 256 + c) * KTJ;
  const float* tp = tem1 + c * KTJ;
  float ofc = o34[c] * (1.f / 3.f);
#pragma unroll
  for (int j = 0; j < 25; ++j) {
    int node = j * 20 + t;
    float lf = ap[node] * (1.f / 3.f) + ofc + tp[node];
    best = fmaxf(best, lf);
  }
  y0[((long)b * 256 + c) * 20 + t] = best;
}
// temporal conv k=3 + bn1 + relu
__global__ void sgn_cnn1(const float* __restrict__ y0, const float* __restrict__ w,
                         const float* __restrict__ bb, const float* __restrict__ sc,
                         const float* __restrict__ of, float* __restrict__ y1) {
  int i = blockIdx.x * 256 + threadIdx.x;
  if (i >= KBS * 256 * 20) return;
  int t = i % 20, o = (i / 20) % 256, b = i / (256 * 20);
  float a = bb[o];
  for (int c = 0; c < 256; ++c) {
    const float* wp = &w[((long)o * 256 + c) * 3];
    const float* xp = &y0[((long)b * 256 + c) * 20];
    if (t > 0)  a += wp[0] * xp[t - 1];
    a += wp[1] * xp[t];
    if (t < 19) a += wp[2] * xp[t + 1];
  }
  a = a * sc[o] + of[o];
  y1[((long)b * 256 + o) * 20 + t] = fmaxf(a, 0.f);
}
// 1x1 conv 256->512 + bn2 + relu
__global__ void sgn_cnn2(const float* __restrict__ y1, const float* __restrict__ w,
                         const float* __restrict__ bb, const float* __restrict__ sc,
                         const float* __restrict__ of, float* __restrict__ y2) {
  int i = blockIdx.x * 256 + threadIdx.x;
  if (i >= KBS * 512 * 20) return;
  int t = i % 20, o = (i / 20) % 512, b = i / (512 * 20);
  float a = bb[o];
  for (int c = 0; c < 256; ++c)
    a += w[(long)o * 256 + c] * y1[((long)b * 256 + c) * 20 + t];
  a = a * sc[o] + of[o];
  y2[((long)b * 512 + o) * 20 + t] = fmaxf(a, 0.f);
}
__global__ void sgn_pool(const float* __restrict__ y2, float* __restrict__ feat) {
  int i = blockIdx.x * 256 + threadIdx.x;
  if (i >= KBS * 512) return;
  const float* p = y2 + (long)i * 20;
  float m = p[0];
#pragma unroll
  for (int t = 1; t < 20; ++t) m = fmaxf(m, p[t]);
  feat[i] = m;
}
__global__ void sgn_fc(const float* __restrict__ feat, const float* __restrict__ w,
                       const float* __restrict__ bb, float* __restrict__ out) {
  int i = blockIdx.x * 256 + threadIdx.x;
  if (i >= KBS * 60) return;
  int k = i % 60, b = i / 60;
  float a = bb[k];
  for (int o = 0; o < 512; ++o) a += w[k * 512 + o] * feat[(long)b * 512 + o];
  out[(long)b * 60 + k] = a;
}

// ---------------------------------------------------------------------------
extern "C" void kernel_launch(void* const* d_in, const int* in_sizes, int n_in,
                              void* d_out, int out_size, void* d_ws, size_t ws_size,
                              hipStream_t stream) {
  (void)in_sizes; (void)n_in; (void)out_size; (void)ws_size;
  // JAX pytree (alphabetical) flattening of setup_inputs():
  // 0:A  1-4:bn1(b,g,m,v) 5-8:bn2 9-12:bn34 13:cnn1_b 14:cnn1_w 15:cnn2_b
  // 16:cnn2_w 17-20:data_bn 21:dd_b1 22:dd_b2 23:dd_w1 24:dd_w2 25:dj_b1
  // 26:dj_b2 27:dj_w1 28:dj_w2 29:fc_b 30:fc_w 31:g1_b 32:g1_w 33:g2_b 34:g2_w
  // gcn1_1{35:b1 36-39:bn 40:w 41:w1} gcn1_2{42,43-46,47,48} gcn1_3{49,50-53,54,55}
  // gcn2_1{56,57-60,61,62} gcn2_2{63,64-67,68,69} gcn2_3{70,71-74,75,76}
  // gcn3_1{77:b1 78:w 79:w1} gcn3_2{80,81,82} gcn3_3{83,84,85}
  // 86:spa_b1 87:spa_b2 88:spa_w1 89:spa_w2 90:tem_b1 91:tem_b2 92:tem_w1
  // 93:tem_w2 94:spa 95:tem 96:x
  auto F = [&](int i) { return (const float*)d_in[i]; };

  size_t off = 0;
  auto alloc = [&](size_t bytes) {
    void* p = (char*)d_ws + off;
    off += (bytes + 255) & ~(size_t)255;
    return p;
  };
  // weights (bf16)
  u16* Abf  = (u16*)alloc(250000 * 2);
  u16* Wg1  = (u16*)alloc(32768 * 2);
  u16* Wg2  = (u16*)alloc(32768 * 2);
  u16* Wc11 = (u16*)alloc(32768 * 2);   // 128 x 256
  u16* Wc12 = (u16*)alloc(65536 * 2);   // 256 x 256
  u16* Wc13 = (u16*)alloc(131072 * 2);  // 256 x 512
  u16* Wc21 = (u16*)alloc(32768 * 2);
  u16* Wc22 = (u16*)alloc(65536 * 2);
  u16* Wc23 = (u16*)alloc(131072 * 2);
  u16* Wc31 = (u16*)alloc(32768 * 2);
  u16* Wc32 = (u16*)alloc(65536 * 2);
  u16* Wc33 = (u16*)alloc(131072 * 2);
  // folded BN (f32)
  float* s11 = (float*)alloc(128 * 4); float* o11 = (float*)alloc(128 * 4);
  float* s12 = (float*)alloc(256 * 4); float* o12 = (float*)alloc(256 * 4);
  float* s13 = (float*)alloc(256 * 4); float* o13 = (float*)alloc(256 * 4);
  float* s21 = (float*)alloc(128 * 4); float* o21 = (float*)alloc(128 * 4);
  float* s22 = (float*)alloc(256 * 4); float* o22 = (float*)alloc(256 * 4);
  float* s23 = (float*)alloc(256 * 4); float* o23 = (float*)alloc(256 * 4);
  float* s34 = (float*)alloc(256 * 4); float* o34 = (float*)alloc(256 * 4);
  float* sb1 = (float*)alloc(256 * 4); float* ob1 = (float*)alloc(256 * 4);
  float* sb2 = (float*)alloc(512 * 4); float* ob2 = (float*)alloc(512 * 4);
  // batch-invariant one-hot embeds
  float* spaT  = (float*)alloc(64 * 500 * 4);
  float* spa1s = (float*)alloc(64 * 500 * 4);
  float* temT  = (float*)alloc(64 * 500 * 4);
  float* tem1s = (float*)alloc(256 * 500 * 4);
  // activations
  float* Lin = (float*)alloc((size_t)256 * 6 * 500 * 4);
  float* dif = (float*)alloc((size_t)256 * 6 * 500 * 4);
  void*  Ereg = alloc((size_t)65536000);             // Epos1+Edif1, later madj
  float* Epos1 = (float*)Ereg;
  float* Edif1 = Epos1 + (size_t)256 * 64 * 500;
  u16*   madj  = (u16*)Ereg;                          // 256*256*500 bf16
  u16* Linput = (u16*)alloc((size_t)256 * 128 * 500 * 2);
  u16* g1b = (u16*)alloc((size_t)256 * 256 * 500 * 2);  // later y_a
  u16* g2b = (u16*)alloc((size_t)256 * 256 * 500 * 2);  // later y_b
  u16* LG  = (u16*)alloc((size_t)256 * 500 * 500 * 2);
  float* accum = (float*)alloc((size_t)256 * 256 * 500 * 4);
  float* y0 = (float*)alloc((size_t)256 * 256 * 20 * 4);
  float* y1 = (float*)alloc((size_t)256 * 256 * 20 * 4);
  float* y2 = (float*)alloc((size_t)256 * 512 * 20 * 4);
  float* feat = (float*)alloc((size_t)256 * 512 * 4);
  u16* y_a = g1b;
  u16* y_b = g2b;

  auto g1d = [](int n) { return dim3((n + 255) / 256); };

  // ---- weight conversion / packing / BN folding ----
  sgn_f2bf<<<g1d(250000), 256, 0, stream>>>(F(0), Abf, 250000);
  sgn_f2bf<<<g1d(32768), 256, 0, stream>>>(F(32), Wg1, 32768);
  sgn_f2bf<<<g1d(32768), 256, 0, stream>>>(F(34), Wg2, 32768);
  sgn_pack<<<g1d(128 * 128), 256, 0, stream>>>(F(40), F(41), Wc11, 128, 128);
  sgn_pack<<<g1d(256 * 128), 256, 0, stream>>>(F(47), F(48), Wc12, 256, 128);
  sgn_pack<<<g1d(256 * 256), 256, 0, stream>>>(F(54), F(55), Wc13, 256, 256);
  sgn_pack<<<g1d(128 * 128), 256, 0, stream>>>(F(61), F(62), Wc21, 128, 128);
  sgn_pack<<<g1d(256 * 128), 256, 0, stream>>>(F(68), F(69), Wc22, 256, 128);
  sgn_pack<<<g1d(256 * 256), 256, 0, stream>>>(F(75), F(76), Wc23, 256, 256);
  sgn_pack<<<g1d(128 * 128), 256, 0, stream>>>(F(78), F(79), Wc31, 128, 128);
  sgn_pack<<<g1d(256 * 128), 256, 0, stream>>>(F(81), F(82), Wc32, 256, 128);
  sgn_pack<<<g1d(256 * 256), 256, 0, stream>>>(F(84), F(85), Wc33, 256, 256);
  sgn_bnfold<<<2, 64, 0, stream>>>(F(37), F(36), F(38), F(39), 128, s11, o11);
  sgn_bnfold<<<4, 64, 0, stream>>>(F(44), F(43), F(45), F(46), 256, s12, o12);
  sgn_bnfold<<<4, 64, 0, stream>>>(F(51), F(50), F(52), F(53), 256, s13, o13);
  sgn_bnfold<<<2, 64, 0, stream>>>(F(58), F(57), F(59), F(60), 128, s21, o21);
  sgn_bnfold<<<4, 64, 0, stream>>>(F(65), F(64), F(66), F(67), 256, s22, o22);
  sgn_bnfold<<<4, 64, 0, stream>>>(F(72), F(71), F(73), F(74), 256, s23, o23);
  sgn_bnfold<<<4, 64, 0, stream>>>(F(10), F(9), F(11), F(12), 256, s34, o34);
  sgn_bnfold<<<4, 64, 0, stream>>>(F(2), F(1), F(3), F(4), 256, sb1, ob1);
  sgn_bnfold<<<8, 64, 0, stream>>>(F(6), F(5), F(7), F(8), 512, sb2, ob2);

  // ---- batch-invariant spa/tem embeds (one-hot first layer = lookup) ----
  sgn_onehot<<<g1d(64 * 500), 256, 0, stream>>>(F(88), F(86), spaT, 25, 0);
  sgn_dense2<<<g1d(64 * 500), 256, 0, stream>>>(spaT, F(89), F(87), spa1s, 64);
  sgn_onehot<<<g1d(64 * 500), 256, 0, stream>>>(F(92), F(90), temT, 20, 1);
  sgn_dense2<<<g1d(256 * 500), 256, 0, stream>>>(temT, F(93), F(91), tem1s, 256);

  // ---- data_bn + diff + joint/diff embeds -> Linput (bf16) ----
  sgn_prep<<<g1d(256 * 500 * 6), 256, 0, stream>>>(F(96), F(17), F(18), F(19), F(20), Lin, dif);
  sgn_embed1<<<g1d(256 * 64 * 500), 256, 0, stream>>>(Lin, F(27), F(25), Epos1);
  sgn_embed1<<<g1d(256 * 64 * 500), 256, 0, stream>>>(dif, F(23), F(21), Edif1);
  sgn_linput<<<g1d(256 * 64 * 500), 256, 0, stream>>>(Epos1, Edif1, F(28), F(26), F(24), F(22), Linput);
  sgn_bcast<<<g1d(256 * 64 * 500), 256, 0, stream>>>(spa1s, Linput);

  auto gemm = [&](const u16* A, long a_bs, int sam, int sak,
                  const u16* B0, long b0_bs, const u16* B1, long b1_bs,
                  int sbk, int sbn, int K1, int K, int M, int N,
                  const float* bias, const float* sc, const float* of, const float* chs,
                  float alpha, int em, u16* outb, float* outf, long o_bs, int som, int son) {
    dim3 g((N + 127) / 128, (M + 63) / 64, KBS);
    sgn_gemm<<<g, 256, 0, stream>>>(A, a_bs, sam, sak, B0, b0_bs, B1, b1_bs,
                                    sbk, sbn, K1, K, M, N, bias, sc, of, chs,
                                    alpha, em, outb, outf, o_bs, som, son);
  };

  // ---- learned adjacency: g1, g2 node-major [node][256]; LG = softmax(g1 g2^T)
  gemm(Wg1, 0, 128, 1, Linput, 64000, Linput, 64000, 500, 1, 128, 128,
       256, 500, F(31), 0, 0, 0, 1.f, EM_BIAS, g1b, 0, 128000, 1, 256);
  gemm(Wg2, 0, 128, 1, Linput, 64000, Linput, 64000, 500, 1, 128, 128,
       256, 500, F(33), 0, 0, 0, 1.f, EM_BIAS, g2b, 0, 128000, 1, 256);
  gemm(g1b, 128000, 256, 1, g2b, 128000, g2b, 128000, 1, 256, 256, 256,
       500, 500, 0, 0, 0, 0, 1.f, 0, LG, 0, 250000, 500, 1);
  sgn_softmax<<<dim3(KTJ, KBS), 256, 0, stream>>>(LG);

  // ---- four GCN branches, streamed into one f32 accumulator ----
  sgn_zero<<<g1d(256 * 256 * 500), 256, 0, stream>>>(accum, 256 * 256 * 500);

  const u16* Gp[4]  = { LG, Abf, Abf, LG };
  long Gbs[4]       = { 250000, 0, 0, 250000 };
  const u16* Wl[4][3] = { {Wc11, Wc12, Wc13}, {Wc21, Wc22, Wc23},
                          {Wc31, Wc32, Wc33}, {Wc31, Wc32, Wc33} };
  const float* Bl[4][3] = { {F(35), F(42), F(49)}, {F(56), F(63), F(70)},
                            {F(77), F(80), F(83)}, {F(77), F(80), F(83)} };
  const float* Sl[4][3] = { {s11, s12, s13}, {s21, s22, s23}, {0, 0, 0}, {0, 0, 0} };
  const float* Ol[4][3] = { {o11, o12, o13}, {o21, o22, o23}, {0, 0, 0}, {0, 0, 0} };
  const int useBN[4] = { 1, 1, 0, 0 };
  const int cin[3] = { 128, 128, 256 }, cout[3] = { 128, 256, 256 };

  for (int br = 0; br < 4; ++br) {
    const u16* xin = Linput;
    long xin_bs = 64000;
    u16* ybuf[2] = { y_a, y_b };
    for (int l = 0; l < 3; ++l) {
      int ci = cin[l], co = cout[l];
      // madj[c,n] = sum_m G[n,m] x[c,m]  (A = x, B = G with sbk==1 fast path)
      gemm(xin, xin_bs, 500, 1, Gp[br], Gbs[br], Gp[br], Gbs[br], 1, 500,
           500, 500, ci, 500, 0, 0, 0, 0, 1.f, 0, madj, 0, (long)ci * 500, 500, 1);
      if (l < 2) {
        int em = useBN[br] ? (EM_BIAS | EM_BN | EM_RELU) : (EM_BIAS | EM_RELU);
        gemm(Wl[br][l], 0, 2 * ci, 1, madj, (long)ci * 500, xin, xin_bs, 500, 1,
             ci, 2 * ci, co, 500, Bl[br][l], Sl[br][l], Ol[br][l], 0, 1.f, em,
             ybuf[l], 0, (long)co * 500, 500, 1);
        xin = ybuf[l];
        xin_bs = (long)co * 500;
      } else {
        int em = (useBN[br] ? (EM_BIAS | EM_BN | EM_RELU) : (EM_BIAS | EM_RELU)) | EM_ACC;
        float alpha = useBN[br] ? 1.f : 0.5f;           // i34 = 0.5*(i3+i4)*s34 + o34
        const float* chs = useBN[br] ? nullptr : s34;
        gemm(Wl[br][l], 0, 2 * ci, 1, madj, (long)ci * 500, xin, xin_bs, 500, 1,
             ci, 2 * ci, co, 500, Bl[br][l], Sl[br][l], Ol[br][l], chs, alpha, em,
             0, accum, 128000, 500, 1);
      }
    }
  }

  // ---- combine + joint max + temporal head ----
  sgn_combmax<<<g1d(256 * 256 * 20), 256, 0, stream>>>(accum, o34, tem1s, y0);
  sgn_cnn1<<<g1d(256 * 256 * 20), 256, 0, stream>>>(y0, F(14), F(13), sb1, ob1, y1);
  sgn_cnn2<<<g1d(256 * 512 * 20), 256, 0, stream>>>(y1, F(16), F(15), sb2, ob2, y2);
  sgn_pool<<<g1d(256 * 512), 256, 0, stream>>>(y2, feat);
  sgn_fc<<<g1d(256 * 60), 256, 0, stream>>>(feat, F(30), F(29), (float*)d_out);
}
